// SimplifiedMamba_59012850647637
// MI455X (gfx1250) — matrix operand order repaired and checked by reading, not compiled, not run on hardware
//
#include <hip/hip_runtime.h>
#include <math.h>

// ---------------------------------------------------------------------------
// SimplifiedMamba forward for MI455X (gfx1250, wave32, WMMA).
// GEMMs run on v_wmma_f32_16x16x32_bf16 with 32x64-per-wave blocking;
// SSM scan is register-resident per channel with LDS broadcast of B/C;
// xp staging uses GLOBAL_LOAD_ASYNC_TO_LDS (ASYNCcnt) when available.
// ---------------------------------------------------------------------------

constexpr int kV  = 32000;
constexpr int kD  = 1024;
constexpr int kNL = 4;
constexpr int kN  = 16;
constexpr int kDI = 2048;   // D * E
constexpr int kL  = 1024;
constexpr int kXP = 2 * kN + 1;  // 33

typedef __attribute__((ext_vector_type(16))) __bf16 v16bf;
typedef __attribute__((ext_vector_type(8)))  __bf16 v8bf;
typedef __attribute__((ext_vector_type(8)))  float  v8f;
typedef int v4i_vs __attribute__((vector_size(16)));  // builtin's pointee type

#if defined(__has_builtin)
#if __has_builtin(__builtin_amdgcn_global_load_async_to_lds_b128) && \
    __has_builtin(__builtin_amdgcn_s_wait_asynccnt)
#define USE_ASYNC_LDS 1
#endif
#endif

// ---------------------------------------------------------------------------
// Weight prep: transpose (R x C f32) -> (C x R bf16), grid.z = layer.
// ---------------------------------------------------------------------------
__global__ void k_transpose_bf16(const float* __restrict__ in,
                                 __bf16* __restrict__ out, int R, int C) {
  const float* src = in  + (size_t)blockIdx.z * R * C;
  __bf16*      dst = out + (size_t)blockIdx.z * R * C;
  __shared__ float tile[32][33];
  int c0 = blockIdx.x * 32, r0 = blockIdx.y * 32;
#pragma unroll
  for (int i = 0; i < 4; ++i) {
    int r = r0 + threadIdx.y + i * 8;
    tile[threadIdx.y + i * 8][threadIdx.x] =
        src[(size_t)r * C + c0 + threadIdx.x];
  }
  __syncthreads();
#pragma unroll
  for (int i = 0; i < 4; ++i) {
    int c = c0 + threadIdx.y + i * 8;
    dst[(size_t)c * R + r0 + threadIdx.x] =
        (__bf16)tile[threadIdx.x][threadIdx.y + i * 8];
  }
}

// ---------------------------------------------------------------------------
// Embedding gather: x[l, :] = emb[tokens[l], :]
// ---------------------------------------------------------------------------
__global__ void k_embed(const int* __restrict__ tok,
                        const float* __restrict__ emb,
                        float* __restrict__ x) {
  int l = blockIdx.x;
  int t = tok[l];
  for (int d = threadIdx.x; d < kD; d += blockDim.x)
    x[(size_t)l * kD + d] = emb[(size_t)t * kD + d];
}

// ---------------------------------------------------------------------------
// RMSNorm row kernel, bf16 output (feeds WMMA A operand).
// ---------------------------------------------------------------------------
__global__ void k_rmsnorm(const float* __restrict__ x,
                          const float* __restrict__ w,
                          __bf16* __restrict__ out, int Dd) {
  int l = blockIdx.x;
  const float* xr = x + (size_t)l * Dd;
  __shared__ float sd[256];
  float ss = 0.f;
  for (int d = threadIdx.x; d < Dd; d += 256) {
    float v = xr[d];
    ss += v * v;
  }
  sd[threadIdx.x] = ss;
  __syncthreads();
  for (int s = 128; s > 0; s >>= 1) {
    if (threadIdx.x < s) sd[threadIdx.x] += sd[threadIdx.x + s];
    __syncthreads();
  }
  float scale = rsqrtf(sd[0] / (float)Dd + 1e-5f);
  for (int d = threadIdx.x; d < Dd; d += 256)
    out[(size_t)l * Dd + d] = (__bf16)(xr[d] * scale * w[d]);
}

// ---------------------------------------------------------------------------
// A-fragment loader: contiguous K order matching the B fragments:
// lanes 0-15 hold K=0..15, lanes 16-31 hold K=16..31.
// ---------------------------------------------------------------------------
__device__ __forceinline__ v16bf load_a_frag(const __bf16* Arow, int k0,
                                             int hf) {
  v8bf alo = *(const v8bf*)(Arow + k0 + hf * 16);
  v8bf ahi = *(const v8bf*)(Arow + k0 + hf * 16 + 8);
  v16bf a;
#pragma unroll
  for (int i = 0; i < 8; ++i) { a[i] = alo[i]; a[i + 8] = ahi[i]; }
  return a;
}

// ---------------------------------------------------------------------------
// WMMA NT GEMM: C[M x Nn] = A[M x K](bf16 rowmajor) * Bt[Nn x K](bf16 rowmajor)
//               + bias[n] (+ resid). One wave computes a 32x64 strip:
// 2 A fragments, 4 B fragments reused across both M-tiles (8 WMMA/K-step).
// ---------------------------------------------------------------------------
__global__ void k_gemm_bb(const __bf16* __restrict__ A,
                          const __bf16* __restrict__ Bt,
                          const float* __restrict__ bias,
                          float* __restrict__ Cout,
                          const float* __restrict__ resid,
                          int M, int Nn, int K) {
  int wid    = blockIdx.x * (blockDim.x >> 5) + (threadIdx.x >> 5);
  int groups = Nn >> 6;                 // 64 columns per wave
  int tm = wid / groups, tg = wid % groups;
  if (tm >= (M >> 5)) return;
  int lane = threadIdx.x & 31;
  int r = lane & 15, hf = lane >> 4;

  const __bf16* Arow0 = A + (size_t)(tm * 32 + r) * K;
  const __bf16* Arow1 = Arow0 + (size_t)16 * K;
  v8f acc[2][4] = {};
  for (int k0 = 0; k0 < K; k0 += 32) {
    v16bf a0 = load_a_frag(Arow0, k0, hf);
    v16bf a1 = load_a_frag(Arow1, k0, hf);
    if (k0 + 64 < K) {
      __builtin_prefetch(Arow0 + k0 + 64, 0, 1);
      __builtin_prefetch(Arow1 + k0 + 64, 0, 1);
    }
#pragma unroll
    for (int g = 0; g < 4; ++g) {
      // B 32x16 fragment: lane n=r holds 16 contiguous K (hf selects half).
      v16bf b = *(const v16bf*)(Bt + (size_t)(tg * 64 + g * 16 + r) * K +
                                k0 + hf * 16);
      acc[0][g] = __builtin_amdgcn_wmma_f32_16x16x32_bf16(
          false, a0, false, b, (short)0, acc[0][g], false, false);
      acc[1][g] = __builtin_amdgcn_wmma_f32_16x16x32_bf16(
          false, a1, false, b, (short)0, acc[1][g], false, false);
    }
  }
#pragma unroll
  for (int m = 0; m < 2; ++m) {
#pragma unroll
    for (int g = 0; g < 4; ++g) {
      int n = tg * 64 + g * 16 + r;
      float bi = bias ? bias[n] : 0.f;
#pragma unroll
      for (int j = 0; j < 8; ++j) {
        size_t idx = (size_t)(tm * 32 + m * 16 + hf * 8 + j) * Nn + n;
        float v = acc[m][g][j] + bi;
        if (resid) v += resid[idx];
        Cout[idx] = v;
      }
    }
  }
}

// Head GEMM: B operand is f32 (emb rows), converted to bf16 in-register.
// Same 32x64 blocking so each converted B fragment feeds two WMMAs.
__global__ void k_gemm_bf(const __bf16* __restrict__ A,
                          const float* __restrict__ Bt,
                          const float* __restrict__ bias,
                          float* __restrict__ Cout,
                          int M, int Nn, int K) {
  int wid    = blockIdx.x * (blockDim.x >> 5) + (threadIdx.x >> 5);
  int groups = Nn >> 6;
  int tm = wid / groups, tg = wid % groups;
  if (tm >= (M >> 5)) return;
  int lane = threadIdx.x & 31;
  int r = lane & 15, hf = lane >> 4;

  const __bf16* Arow0 = A + (size_t)(tm * 32 + r) * K;
  const __bf16* Arow1 = Arow0 + (size_t)16 * K;
  v8f acc[2][4] = {};
  for (int k0 = 0; k0 < K; k0 += 32) {
    v16bf a0 = load_a_frag(Arow0, k0, hf);
    v16bf a1 = load_a_frag(Arow1, k0, hf);
#pragma unroll
    for (int g = 0; g < 4; ++g) {
      const float* Brow =
          Bt + (size_t)(tg * 64 + g * 16 + r) * K + k0 + hf * 16;
      if (k0 + 32 < K) __builtin_prefetch(Brow + 32, 0, 1);
      v16bf b;
#pragma unroll
      for (int i = 0; i < 16; ++i) b[i] = (__bf16)Brow[i];
      acc[0][g] = __builtin_amdgcn_wmma_f32_16x16x32_bf16(
          false, a0, false, b, (short)0, acc[0][g], false, false);
      acc[1][g] = __builtin_amdgcn_wmma_f32_16x16x32_bf16(
          false, a1, false, b, (short)0, acc[1][g], false, false);
    }
  }
#pragma unroll
  for (int m = 0; m < 2; ++m) {
#pragma unroll
    for (int g = 0; g < 4; ++g) {
      int n = tg * 64 + g * 16 + r;
      float bi = bias ? bias[n] : 0.f;
#pragma unroll
      for (int j = 0; j < 8; ++j) {
        size_t idx = (size_t)(tm * 32 + m * 16 + hf * 8 + j) * Nn + n;
        Cout[idx] = acc[m][g][j] + bi;
      }
    }
  }
}

// ---------------------------------------------------------------------------
// xi = silu(xr[:, :DI])
// ---------------------------------------------------------------------------
__global__ void k_silu_split(const float* __restrict__ xr,
                             float* __restrict__ xi) {
  size_t i = (size_t)blockIdx.x * blockDim.x + threadIdx.x;
  int l = (int)(i / kDI), d = (int)(i % kDI);
  float v = xr[(size_t)l * (2 * kDI) + d];
  xi[i] = v / (1.f + expf(-v));
}

// ---------------------------------------------------------------------------
// xp[l, 0:33] = xi[l, :] @ xp_w + xp_b.  N=33 is too skinny for WMMA:
// stage the row into LDS (async DMA path when available), then one thread
// per output column (coalesced xp_w reads, LDS broadcast reads).
// ---------------------------------------------------------------------------
__global__ void k_xp(const float* __restrict__ xi,
                     const float* __restrict__ xpw,
                     const float* __restrict__ xpb,
                     float* __restrict__ xp) {
  int l = blockIdx.x;
  __shared__ float row[kDI];
#if USE_ASYNC_LDS
  // 16B per issue, tracked on ASYNCcnt; no VGPR round-trip.
  for (int d = threadIdx.x * 4; d < kDI; d += blockDim.x * 4)
    __builtin_amdgcn_global_load_async_to_lds_b128(
        (__attribute__((address_space(1))) v4i_vs*)(xi + (size_t)l * kDI + d),
        (__attribute__((address_space(3))) v4i_vs*)&row[d], 0, 0);
  __builtin_amdgcn_s_wait_asynccnt(0);
  __syncthreads();
#else
  for (int d = threadIdx.x; d < kDI; d += blockDim.x)
    row[d] = xi[(size_t)l * kDI + d];
  __syncthreads();
#endif
  int j = threadIdx.x;
  if (j < kXP) {
    float acc = xpb[j];
    for (int d = 0; d < kDI; ++d) acc += row[d] * xpw[d * kXP + j];
    xp[(size_t)l * kXP + j] = acc;
  }
}

// ---------------------------------------------------------------------------
// Selective-scan recurrence: sequential over L, parallel over DI channels.
// Each thread owns one channel: 16-element state + A row in registers,
// per-timestep B/C/delta-scalar broadcast through LDS.
// ---------------------------------------------------------------------------
__global__ void k_scan(const float* __restrict__ xp,
                       const float* __restrict__ xi,
                       const float* __restrict__ dtw_g,
                       const float* __restrict__ dtb_g,
                       const float* __restrict__ A_log,
                       const float* __restrict__ D_p,
                       float* __restrict__ y) {
  int d = blockIdx.x * blockDim.x + threadIdx.x;  // < kDI
  __shared__ float sxp[kXP];
  float Arow[kN];
#pragma unroll
  for (int n = 0; n < kN; ++n) Arow[n] = -expf(A_log[(size_t)d * kN + n]);
  float dtw = dtw_g[d], dtb = dtb_g[d];
  float Dp = fminf(fmaxf(D_p[d], -100.f), 100.f);
  float st[kN];
#pragma unroll
  for (int n = 0; n < kN; ++n) st[n] = 0.f;

  for (int t = 0; t < kL; ++t) {
    if (threadIdx.x < kXP) sxp[threadIdx.x] = xp[(size_t)t * kXP + threadIdx.x];
    __syncthreads();
    float z = sxp[0] * dtw + dtb;
    float delta = (z > 20.f ? z : log1pf(expf(z))) + 1e-8f;
    float xt = xi[(size_t)t * kDI + d];
    float yacc = 0.f;
#pragma unroll
    for (int n = 0; n < kN; ++n) {
      float dA = expf(delta * Arow[n]);
      dA = fminf(fmaxf(dA, 1e-12f), 1e4f);
      float dBu = delta * sxp[1 + n] * xt;
      dBu = fminf(fmaxf(dBu, -1e4f), 1e4f);
      float s = dA * st[n] + dBu;
      s = fminf(fmaxf(s, -1e6f), 1e6f);
      st[n] = s;
      yacc += s * sxp[1 + kN + n];
    }
    yacc = fminf(fmaxf(yacc, -1e6f), 1e6f);
    y[(size_t)t * kDI + d] = yacc + xt * Dp;
    __syncthreads();  // protect sxp before next timestep's reload
  }
}

// ---------------------------------------------------------------------------
// yg = bf16( y * silu(res) ), res = xr[:, DI:]
// ---------------------------------------------------------------------------
__global__ void k_gate(const float* __restrict__ y,
                       const float* __restrict__ xr,
                       __bf16* __restrict__ yg) {
  size_t i = (size_t)blockIdx.x * blockDim.x + threadIdx.x;
  int l = (int)(i / kDI), d = (int)(i % kDI);
  float rv = xr[(size_t)l * (2 * kDI) + kDI + d];
  float s = rv / (1.f + expf(-rv));
  yg[i] = (__bf16)(y[i] * s);
}

// ---------------------------------------------------------------------------
// Host-side orchestration.
// ---------------------------------------------------------------------------
extern "C" void kernel_launch(void* const* d_in, const int* in_sizes, int n_in,
                              void* d_out, int out_size, void* d_ws,
                              size_t ws_size, hipStream_t stream) {
  (void)in_sizes; (void)n_in; (void)out_size; (void)ws_size;
  const int*   tokens   = (const int*)d_in[0];
  const float* emb      = (const float*)d_in[1];
  const float* head_b   = (const float*)d_in[2];
  const float* norm_f_w = (const float*)d_in[3];
  const float* in_w     = (const float*)d_in[4];
  const float* in_b     = (const float*)d_in[5];
  const float* out_w    = (const float*)d_in[6];
  const float* out_b    = (const float*)d_in[7];
  const float* xp_w     = (const float*)d_in[8];
  const float* xp_b     = (const float*)d_in[9];
  const float* dt_w     = (const float*)d_in[10];
  const float* dt_b     = (const float*)d_in[11];
  const float* A_log    = (const float*)d_in[12];
  const float* D_p      = (const float*)d_in[13];
  const float* norm_w   = (const float*)d_in[14];

  char* wsb = (char*)d_ws;
  size_t off = 0;
  auto take = [&](size_t bytes) -> char* {
    char* p = wsb + off;
    off = (off + bytes + 255) & ~(size_t)255;
    return p;
  };
  float*  x   = (float*)  take((size_t)kL * kD * 4);
  __bf16* h   = (__bf16*) take((size_t)kL * kD * 2);
  __bf16* xf  = (__bf16*) take((size_t)kL * kD * 2);
  __bf16* w1t = (__bf16*) take((size_t)kNL * 2 * kDI * kD * 2);
  __bf16* w2t = (__bf16*) take((size_t)kNL * kDI * kD * 2);
  float*  xr  = (float*)  take((size_t)kL * 2 * kDI * 4);
  float*  xi  = (float*)  take((size_t)kL * kDI * 4);
  float*  xp  = (float*)  take((size_t)kL * kXP * 4);
  float*  y   = (float*)  take((size_t)kL * kDI * 4);
  __bf16* yg  = (__bf16*) take((size_t)kL * kDI * 2);

  // Weight prep (bf16, N-major so all GEMMs are NT with contiguous K).
  k_transpose_bf16<<<dim3(2 * kDI / 32, kD / 32, kNL), dim3(32, 8), 0, stream>>>(
      in_w, w1t, kD, 2 * kDI);
  k_transpose_bf16<<<dim3(kD / 32, kDI / 32, kNL), dim3(32, 8), 0, stream>>>(
      out_w, w2t, kDI, kD);

  k_embed<<<kL, 256, 0, stream>>>(tokens, emb, x);

  const int ew_grid = (kL * kDI) / 256;
  for (int lyr = 0; lyr < kNL; ++lyr) {
    k_rmsnorm<<<kL, 256, 0, stream>>>(x, norm_w + (size_t)lyr * kD, h, kD);

    // xr = h @ in_w + in_b   (M=1024, N=4096, K=1024)
    int blocks1 = (kL / 32) * (2 * kDI / 64) / 8;
    k_gemm_bb<<<blocks1, 256, 0, stream>>>(
        h, w1t + (size_t)lyr * 2 * kDI * kD, in_b + (size_t)lyr * 2 * kDI,
        xr, nullptr, kL, 2 * kDI, kD);

    k_silu_split<<<ew_grid, 256, 0, stream>>>(xr, xi);
    k_xp<<<kL, 64, 0, stream>>>(xi, xp_w + (size_t)lyr * kDI * kXP,
                                xp_b + (size_t)lyr * kXP, xp);
    k_scan<<<kDI / 256, 256, 0, stream>>>(
        xp, xi, dt_w + (size_t)lyr * kDI, dt_b + (size_t)lyr * kDI,
        A_log + (size_t)lyr * kDI * kN, D_p + (size_t)lyr * kDI, y);
    k_gate<<<ew_grid, 256, 0, stream>>>(y, xr, yg);

    // x = x + yg @ out_w + out_b   (M=1024, N=1024, K=2048), residual fused
    int blocks2 = (kL / 32) * (kD / 64) / 8;
    k_gemm_bb<<<blocks2, 256, 0, stream>>>(
        yg, w2t + (size_t)lyr * kDI * kD, out_b + (size_t)lyr * kD,
        x, x, kL, kD, kDI);
  }

  k_rmsnorm<<<kL, 256, 0, stream>>>(x, norm_f_w, xf, kD);

  // logits = xf @ emb.T + head_b   (M=1024, N=32000, K=1024)
  int blocksH = (kL / 32) * (kV / 64) / 8;
  k_gemm_bf<<<blocksH, 256, 0, stream>>>(xf, emb, head_b, (float*)d_out,
                                         kL, kV, kD);
}
